// TemporalAxialAttention_29764123361421
// MI455X (gfx1250) — matrix operand order repaired and checked by reading, not compiled, not run on hardware
//
#include <hip/hip_runtime.h>
#include <hip/hip_fp16.h>

// ---------------- CDNA5 WMMA types ----------------
typedef __attribute__((ext_vector_type(16))) _Float16     v16h;
typedef __attribute__((ext_vector_type(8)))  float        v8f;
typedef __attribute__((ext_vector_type(4)))  unsigned int v4u;
typedef __attribute__((ext_vector_type(4)))  float        v4f;

union AFrag { v16h v; _Float16 h[16]; v4u q[2]; };
union AF32  { v4f  v[4]; float f[16]; };
union CFrag { v8f  v; float f[8]; };

#define NTOK         65536    // B*T*H*W
#define D_MODEL      512
#define E_QKV        1536
#define HEADS        8
#define DHEAD        64
#define SEQLEN       32
#define TOK_T_STRIDE 1024     // H*W (token stride along t)

static __device__ __forceinline__ v8f wmma_f16(const AFrag& a, const AFrag& b, v8f c) {
  // D = A(16x32 f16) * B(32x16 f16) + C(16x16 f32)
  return __builtin_amdgcn_wmma_f32_16x16x32_f16(false, a.v, false, b.v,
                                                (short)0, c, false, false);
}

// ---------------- Kernel 0: weight fp32 -> fp16 transpose (tiny, 2 MB) -----
__global__ __launch_bounds__(256)
void convert_weights(const float* __restrict__ wqkv, const float* __restrict__ wout,
                     _Float16* __restrict__ wqkvT, _Float16* __restrict__ woutT) {
  size_t i = (size_t)blockIdx.x * blockDim.x + threadIdx.x;   // 786,432 threads
  if (i < (size_t)D_MODEL * E_QKV) {            // wqkvT[e][d] = wqkv[d][e]
    size_t e = i / D_MODEL, d = i - e * D_MODEL;
    wqkvT[i] = (_Float16)wqkv[d * E_QKV + e];
  }
  if (i < (size_t)D_MODEL * D_MODEL) {          // woutT[d][e] = wout[e][d]
    size_t dd = i / D_MODEL, e = i - dd * D_MODEL;
    woutT[i] = (_Float16)wout[e * D_MODEL + dd];
  }
}

// ---------------- GEMM: C[M,N] = A[M,K] * BT[N,K]^T ------------------------
// One wave computes a 16(M) x 64(N) strip; fragments loaded straight from
// global in WMMA register layout. A may be fp32 (converted in-register) --
// x (134 MB) and all weights are L2-resident on MI455X (192 MB L2), so the
// repeated fragment reads across N-strips never touch HBM twice.
template<int M, int N, int K, bool AF32IN, bool F32OUT>
__global__ __launch_bounds__(256)
void gemm_wmma(const void* __restrict__ Ain, const _Float16* __restrict__ BT,
               _Float16* __restrict__ outH, float* __restrict__ outF,
               const float* __restrict__ bias) {
  int tid     = blockIdx.x * 256 + threadIdx.x;
  int wave    = tid >> 5;
  int lane    = threadIdx.x & 31;
  int lane_lo = lane & 15;
  int hi      = lane >> 4;
  constexpr int nStrips = N >> 6;
  int mTile   = wave / nStrips;
  int nStrip  = wave - mTile * nStrips;
  if (mTile * 16 >= M) return;
  int m0 = mTile * 16;
  int n0 = nStrip * 64;

  CFrag c[4];
  #pragma unroll
  for (int i = 0; i < 4; ++i)
    #pragma unroll
    for (int j = 0; j < 8; ++j) c[i].f[j] = 0.0f;

  const float*    aRowF = (const float*)Ain    + (size_t)(m0 + lane_lo) * K;
  const _Float16* aRowH = (const _Float16*)Ain + (size_t)(m0 + lane_lo) * K;

  #pragma unroll 4
  for (int k0 = 0; k0 < K; k0 += 32) {
    AFrag a;                                   // A 16x32: K = j + (j>=8)*8 + hi*8
    if constexpr (AF32IN) {
      AF32 af;
      af.v[0] = *(const v4f*)(aRowF + k0 + hi * 8);
      af.v[1] = *(const v4f*)(aRowF + k0 + hi * 8 + 4);
      af.v[2] = *(const v4f*)(aRowF + k0 + hi * 8 + 16);
      af.v[3] = *(const v4f*)(aRowF + k0 + hi * 8 + 20);
      #pragma unroll
      for (int j = 0; j < 16; ++j) a.h[j] = (_Float16)af.f[j];
    } else {
      a.q[0] = *(const v4u*)(aRowH + k0 + hi * 8);
      a.q[1] = *(const v4u*)(aRowH + k0 + hi * 8 + 16);
    }
    #pragma unroll
    for (int i = 0; i < 4; ++i) {              // B 32x16: N = lane_lo, K = j + (j>=8)*8 + hi*8
      const _Float16* bRow = BT + (size_t)(n0 + i * 16 + lane_lo) * K + k0 + hi * 8;
      AFrag b;
      b.q[0] = *(const v4u*)(bRow);
      b.q[1] = *(const v4u*)(bRow + 16);
      c[i].v = wmma_f16(a, b, c[i].v);
    }
  }
  #pragma unroll
  for (int i = 0; i < 4; ++i) {
    int n = n0 + i * 16 + lane_lo;
    #pragma unroll
    for (int r = 0; r < 8; ++r) {              // C/D: M = r + hi*8
      int m = m0 + r + hi * 8;
      if (F32OUT) outF[(size_t)m * N + n] = c[i].f[r] + bias[n];
      else        outH[(size_t)m * N + n] = (_Float16)c[i].f[r];
    }
  }
}

// ---------------- Attention: one wave per (sequence, head) ------------------
__global__ __launch_bounds__(128)
void attn_wmma(const _Float16* __restrict__ qkv, _Float16* __restrict__ attnOut) {
  __shared__ __attribute__((aligned(16))) _Float16 ldsAll[4][7168];
  int lane = threadIdx.x & 31;
  int wv   = threadIdx.x >> 5;
  int gw   = blockIdx.x * 4 + wv;              // 0 .. 16383
  int head = gw & (HEADS - 1);
  int seq  = gw >> 3;                          // 0 .. 2047 == b*1024 + h*32 + w
  int b    = seq >> 10;
  int hw   = seq & 1023;
  size_t tokBase = (size_t)b * 32768 + hw;     // token(t) = tokBase + t*1024

  _Float16* qs = ldsAll[wv];                   // [32][64]
  _Float16* ks = qs + 2048;                    // [32][64]
  _Float16* vT = ks + 2048;                    // [64][32] (transposed)
  _Float16* at = vT + 2048;                    // [32][32] attn probs

  // ---- stage q,k (RoPE on first 32 dims) and v (transposed); lane == t ----
  {
    int t = lane;
    const _Float16* qg = qkv + (tokBase + (size_t)t * TOK_T_STRIDE) * E_QKV + head * DHEAD;
    const _Float16* kg = qg + 512;
    const _Float16* vg = qg + 1024;
    #pragma unroll
    for (int i = 0; i < 16; ++i) {
      float inv = __expf(-((float)(2 * i) * (1.0f / 32.0f)) * 9.2103403719761836f); // theta^-(2i/32)
      float ang = (float)t * inv;
      float cs = __cosf(ang), sn = __sinf(ang);
      float q0 = (float)qg[2 * i], q1 = (float)qg[2 * i + 1];
      qs[t * DHEAD + 2 * i]     = (_Float16)(q0 * cs - q1 * sn);
      qs[t * DHEAD + 2 * i + 1] = (_Float16)(q0 * sn + q1 * cs);
      float k0 = (float)kg[2 * i], k1 = (float)kg[2 * i + 1];
      ks[t * DHEAD + 2 * i]     = (_Float16)(k0 * cs - k1 * sn);
      ks[t * DHEAD + 2 * i + 1] = (_Float16)(k0 * sn + k1 * cs);
    }
    #pragma unroll
    for (int d = 32; d < 64; ++d) { qs[t * DHEAD + d] = qg[d]; ks[t * DHEAD + d] = kg[d]; }
    #pragma unroll
    for (int d = 0; d < 64; ++d)  vT[d * 32 + t] = vg[d];
  }
  __syncthreads();

  int lane_lo = lane & 15;
  int hi      = lane >> 4;

  // ---- S = q * k^T  (2x2 tiles of 16x16, K=64 in two 32-steps) ----
  CFrag s[2][2];
  #pragma unroll
  for (int a0 = 0; a0 < 2; ++a0)
    #pragma unroll
    for (int b0 = 0; b0 < 2; ++b0)
      #pragma unroll
      for (int j = 0; j < 8; ++j) s[a0][b0].f[j] = 0.0f;

  #pragma unroll
  for (int ksI = 0; ksI < 2; ++ksI) {
    #pragma unroll
    for (int qi = 0; qi < 2; ++qi) {
      AFrag a;
      const _Float16* ar = qs + (qi * 16 + lane_lo) * DHEAD + ksI * 32 + hi * 8;
      a.q[0] = *(const v4u*)(ar);
      a.q[1] = *(const v4u*)(ar + 16);
      #pragma unroll
      for (int kj = 0; kj < 2; ++kj) {         // B = k^T: lane holds row kj*16+lane_lo of k
        AFrag bb;
        const _Float16* kr = ks + (kj * 16 + lane_lo) * DHEAD + ksI * 32 + hi * 8;
        bb.q[0] = *(const v4u*)(kr);
        bb.q[1] = *(const v4u*)(kr + 16);
        s[qi][kj].v = wmma_f16(a, bb, s[qi][kj].v);
      }
    }
  }

  // ---- causal softmax; rows live at (VGPR r, hi-half), cols across 16 lanes ----
  const float scale = 0.125f;                  // 1/sqrt(64)
  #pragma unroll
  for (int qi = 0; qi < 2; ++qi) {
    #pragma unroll
    for (int r = 0; r < 8; ++r) {
      int tq  = qi * 16 + hi * 8 + r;
      int tk0 = lane_lo, tk1 = 16 + lane_lo;
      float x0 = (tk0 <= tq) ? s[qi][0].f[r] * scale : -1.0e30f;
      float x1 = (tk1 <= tq) ? s[qi][1].f[r] * scale : -1.0e30f;
      float m = fmaxf(x0, x1);
      #pragma unroll
      for (int msk = 1; msk < 16; msk <<= 1) m = fmaxf(m, __shfl_xor(m, msk, 32));
      float p0 = __expf(x0 - m);
      float p1 = __expf(x1 - m);
      float sum = p0 + p1;
      #pragma unroll
      for (int msk = 1; msk < 16; msk <<= 1) sum += __shfl_xor(sum, msk, 32);
      float rinv = 1.0f / sum;
      at[tq * 32 + tk0] = (_Float16)(p0 * rinv);
      at[tq * 32 + tk1] = (_Float16)(p1 * rinv);
    }
  }
  __syncthreads();

  // ---- O = attn * v  (2x4 tiles of 16x16, K=32) ----
  CFrag o[2][4];
  #pragma unroll
  for (int a0 = 0; a0 < 2; ++a0)
    #pragma unroll
    for (int d0 = 0; d0 < 4; ++d0)
      #pragma unroll
      for (int j = 0; j < 8; ++j) o[a0][d0].f[j] = 0.0f;

  #pragma unroll
  for (int qi = 0; qi < 2; ++qi) {
    AFrag a;
    const _Float16* ar = at + (qi * 16 + lane_lo) * 32 + hi * 8;
    a.q[0] = *(const v4u*)(ar);
    a.q[1] = *(const v4u*)(ar + 16);
    #pragma unroll
    for (int dj = 0; dj < 4; ++dj) {           // B = v: lane holds column dj*16+lane_lo via vT
      AFrag bb;
      const _Float16* vr = vT + (dj * 16 + lane_lo) * 32 + hi * 8;
      bb.q[0] = *(const v4u*)(vr);
      bb.q[1] = *(const v4u*)(vr + 16);
      o[qi][dj].v = wmma_f16(a, bb, o[qi][dj].v);
    }
  }

  // ---- scatter back to token-major [n, head*64+d] f16 ----
  #pragma unroll
  for (int qi = 0; qi < 2; ++qi)
    #pragma unroll
    for (int dj = 0; dj < 4; ++dj)
      #pragma unroll
      for (int r = 0; r < 8; ++r) {
        int tq = qi * 16 + hi * 8 + r;
        int d  = dj * 16 + lane_lo;
        attnOut[(tokBase + (size_t)tq * TOK_T_STRIDE) * D_MODEL + head * DHEAD + d] =
            (_Float16)o[qi][dj].f[r];
      }
}

// ---------------- host-side orchestration ----------------------------------
extern "C" void kernel_launch(void* const* d_in, const int* in_sizes, int n_in,
                              void* d_out, int out_size, void* d_ws, size_t ws_size,
                              hipStream_t stream) {
  const float* x     = (const float*)d_in[0];
  const float* wqkv  = (const float*)d_in[1];
  const float* wout  = (const float*)d_in[2];
  const float* bout  = (const float*)d_in[3];
  float*       out   = (float*)d_out;

  _Float16* wqkvT = (_Float16*)d_ws;                              //    786,432 h
  _Float16* woutT = wqkvT + (size_t)D_MODEL * E_QKV;              //    262,144 h
  _Float16* qkvH  = woutT + (size_t)D_MODEL * D_MODEL;            // 100,663,296 h
  _Float16* attnH = qkvH  + (size_t)NTOK * E_QKV;                 // 33,554,432 h
  // total ~270 MB of workspace

  // 1) transpose+convert weights to f16 (B fragments become contiguous-K loads)
  convert_weights<<<3072, 256, 0, stream>>>(wqkv, wout, wqkvT, woutT);

  // 2) QKV projection: [65536 x 512](f32, converted in-register) * [512 x 1536]
  //    waves = (65536/16)*(1536/64) = 98304 -> 12288 blocks of 8 waves
  gemm_wmma<NTOK, E_QKV, D_MODEL, true, false>
      <<<12288, 256, 0, stream>>>(x, wqkvT, qkvH, nullptr, nullptr);

  // 3) causal axial attention with RoPE: 2048 seqs * 8 heads = 16384 waves
  attn_wmma<<<4096, 128, 0, stream>>>(qkvH, attnH);

  // 4) output projection + bias: [65536 x 512] * [512 x 512] -> f32 out
  //    waves = 4096*8 = 32768 -> 4096 blocks
  gemm_wmma<NTOK, D_MODEL, D_MODEL, false, true>
      <<<4096, 256, 0, stream>>>(attnH, woutT, nullptr, out, bout);
}
